// JunctionTreeDensityTransformer_80625126080539
// MI455X (gfx1250) — matrix operand order repaired and checked by reading, not compiled, not run on hardware
//
#include <hip/hip_runtime.h>

typedef __attribute__((ext_vector_type(16))) _Float16 v16h;
typedef __attribute__((ext_vector_type(8)))  _Float16 v8h;
typedef __attribute__((ext_vector_type(8)))  float    v8f;
typedef __attribute__((ext_vector_type(4)))  unsigned int v4u;
typedef __attribute__((ext_vector_type(8)))  int      v8i_;
typedef __attribute__((ext_vector_type(4)))  int      v4i_;

#define LSEQ   1024
#define DMODEL 512
#define NHEAD  8
#define DHEAD  64
#define DFFN   2048
#define NLAYER 6
#define NALPHA 21
#define NTOK   (32 * 1024)   // B*L rows
#define LDSROW 40            // padded LDS tile row stride in halves (80B)

// ---------------- embedding: h = tok_emb[x] + pos_emb (f32) ----------------
__global__ void embed_kernel(const int* __restrict__ x,
                             const float* __restrict__ tok_emb,
                             const float* __restrict__ pos_emb,
                             float* __restrict__ h) {
  int row = blockIdx.x;
  int l   = row & (LSEQ - 1);
  int tok = x[row];
  const float* te = tok_emb + (size_t)tok * DMODEL;
  const float* pe = pos_emb + (size_t)l * DMODEL;
  float* hr = h + (size_t)row * DMODEL;
  for (int d = threadIdx.x; d < DMODEL; d += blockDim.x)
    hr[d] = te[d] + pe[d];
}

// ------------- weight prep: W (K,N) f32 -> Wt (N,K) f16, 32x32 LDS tiles -------------
__global__ __launch_bounds__(256)
void transpose_f16_kernel(const float* __restrict__ W, _Float16* __restrict__ Wt,
                          int K, int N) {
  __shared__ _Float16 ts[32][33];
  int n0 = blockIdx.x * 32, k0 = blockIdx.y * 32;
  int tr = threadIdx.x >> 5, tc = threadIdx.x & 31;
#pragma unroll
  for (int i = 0; i < 4; ++i)
    ts[tr + 8 * i][tc] = (_Float16)W[(size_t)(k0 + tr + 8 * i) * N + (n0 + tc)];
  __syncthreads();
#pragma unroll
  for (int i = 0; i < 4; ++i)
    Wt[(size_t)(n0 + tr + 8 * i) * K + (k0 + tc)] = ts[tc][tr + 8 * i];
}

// ---------------- layernorm: f32 in, f16 out; one wave per row ----------------
__global__ __launch_bounds__(256)
void ln_kernel(const float* __restrict__ in,
               const float* __restrict__ gamma,
               const float* __restrict__ beta,
               _Float16* __restrict__ out) {
  int wid = threadIdx.x >> 5, lane = threadIdx.x & 31;
  int row = blockIdx.x * 8 + wid;
  const float* xr = in + (size_t)row * DMODEL;
  float v[16];
  float s = 0.f, ss = 0.f;
#pragma unroll
  for (int i = 0; i < 16; ++i) {
    v[i] = xr[lane + 32 * i];
    s += v[i]; ss += v[i] * v[i];
  }
#pragma unroll
  for (int o = 16; o; o >>= 1) {
    s  += __shfl_xor(s,  o, 32);
    ss += __shfl_xor(ss, o, 32);
  }
  float mu   = s * (1.0f / DMODEL);
  float var  = ss * (1.0f / DMODEL) - mu * mu;
  float rstd = rsqrtf(var + 1e-6f);
  _Float16* orow = out + (size_t)row * DMODEL;
#pragma unroll
  for (int i = 0; i < 16; ++i) {
    int d = lane + 32 * i;
    orow[d] = (_Float16)((v[i] - mu) * rstd * gamma[d] + beta[d]);
  }
}

// ---------------- TDM tile load: 64 rows x 32 halves, row stride K halves ----------------
// LDS rows padded to 40 halves via D# padding (16 DW interval, 4 DW pad).
__device__ __forceinline__ void tdm_load_tile(unsigned int lds_off,
                                              const _Float16* src,
                                              int row0, int col0,
                                              int nrows_tot, int K) {
  unsigned long long ga =
      (unsigned long long)(uintptr_t)(src + (size_t)row0 * K + col0);
  v4u g0;
  g0[0] = 1u;                                                // count=1, user mode
  g0[1] = lds_off;                                           // lds_addr
  g0[2] = (unsigned int)ga;                                  // global_addr[31:0]
  g0[3] = (unsigned int)((ga >> 32) & 0x1FFFFFFu) | (2u << 30); // addr[56:32] | type=2
  v8i_ g1;
  g1[0] = (int)((1u << 16) |                                 // data_size = 2B
                (1u << 20) |                                 // pad_enable
                (3u << 22) |                                 // pad_interval: 16 DWORDs
                (3u << 25));                                 // pad_amount: 4 DWORDs
  g1[1] = (int)(((unsigned)K & 0xFFFFu) << 16);              // tensor_dim0[15:0]
  g1[2] = (int)(((unsigned)K >> 16) |
                (((unsigned)nrows_tot & 0xFFFFu) << 16));    // dim0 hi | dim1 lo
  g1[3] = (int)(((unsigned)nrows_tot >> 16) | (32u << 16));  // dim1 hi | tile_dim0=32
  g1[4] = 64;                                                // tile_dim1=64, tile_dim2=0
  g1[5] = K;                                                 // tensor_dim0_stride lo32
  g1[6] = 0;
  g1[7] = 0;
  v4i_ z4 = {0, 0, 0, 0};
  v8i_ z8 = {0, 0, 0, 0, 0, 0, 0, 0};
  __builtin_amdgcn_tensor_load_to_lds(g0, g1, z4, z4, z8, 0);
}

// ---------------- fused WMMA GEMM (TDM-staged, double buffered) ----------------
__device__ __forceinline__ float gelu_tanh(float x) {
  float x3 = x * x * x;
  return 0.5f * x * (1.0f + tanhf(0.7978845608028654f * (x + 0.044715f * x3)));
}

// 256 thr = 8 waves; block tile 64(M)x64(N); wave tile 16x32 (2 WMMAs / k-step).
// A: (M,K) f16 row-major.  Wt: (N,K) f16 row-major.
// Output: f16 to Ch (Cf==nullptr) or f32 (+resid) to Cf.
__global__ __launch_bounds__(256)
void gemm_wmma_kernel(const _Float16* __restrict__ A,
                      const _Float16* __restrict__ Wt,
                      const float* __restrict__ bias,
                      const float* resid,
                      float* Cf, _Float16* Ch,
                      int M, int N, int K, int act) {
  __shared__ _Float16 lds[2][2][64 * LDSROW];   // [buf][0=A,1=B]

  const int tid   = threadIdx.x;
  const int wid   = tid >> 5, lane = tid & 31;
  const int waveM = wid >> 1, waveN = wid & 1;
  const int r     = lane & 15, hi = lane >> 4;
  const int m0    = blockIdx.y * 64, n0 = blockIdx.x * 64;
  const int nkt   = K >> 5;

  unsigned int offA0 = (unsigned int)(uintptr_t)&lds[0][0][0];
  unsigned int offB0 = (unsigned int)(uintptr_t)&lds[0][1][0];
  unsigned int offA1 = (unsigned int)(uintptr_t)&lds[1][0][0];
  unsigned int offB1 = (unsigned int)(uintptr_t)&lds[1][1][0];

  v8f acc0 = {};
  v8f acc1 = {};

  if (wid == 0) {                                // stage k-step 0 into buf 0
    tdm_load_tile(offA0, A,  m0, 0, M, K);
    tdm_load_tile(offB0, Wt, n0, 0, N, K);
  }

  for (int kt = 0; kt < nkt; ++kt) {
    const int buf = kt & 1;
    if (kt + 1 < nkt) {
      if (wid == 0) {                            // prefetch next tile pair
        tdm_load_tile(buf ? offA0 : offA1, A,  m0, (kt + 1) << 5, M, K);
        tdm_load_tile(buf ? offB0 : offB1, Wt, n0, (kt + 1) << 5, N, K);
        __builtin_amdgcn_s_wait_tensorcnt(2);    // current pair complete
      }
    } else {
      if (wid == 0) __builtin_amdgcn_s_wait_tensorcnt(0);
    }
    __syncthreads();

    const _Float16* As = &lds[buf][0][0];
    const _Float16* Bs = &lds[buf][1][0];

    // A fragment: lane(row=waveM*16+r): halves K{hi*8..+7} and K{16+hi*8..+7}
    const _Float16* ar = As + (waveM * 16 + r) * LDSROW;
    v8h alo = *(const v8h*)(ar + hi * 8);
    v8h ahi = *(const v8h*)(ar + 16 + hi * 8);
    v16h af = __builtin_shufflevector(alo, ahi, 0, 1, 2, 3, 4, 5, 6, 7,
                                      8, 9, 10, 11, 12, 13, 14, 15);
    // B fragments: lane(col): halves K{hi*8..+7} and K{16+hi*8..+7}
    const _Float16* b0p = Bs + (waveN * 32 + r) * LDSROW;
    v8h b0l = *(const v8h*)(b0p + hi * 8);
    v8h b0h = *(const v8h*)(b0p + 16 + hi * 8);
    v16h bf0 = __builtin_shufflevector(b0l, b0h, 0, 1, 2, 3, 4, 5, 6, 7,
                                       8, 9, 10, 11, 12, 13, 14, 15);
    const _Float16* b1p = b0p + 16 * LDSROW;
    v8h b1l = *(const v8h*)(b1p + hi * 8);
    v8h b1h = *(const v8h*)(b1p + 16 + hi * 8);
    v16h bf1 = __builtin_shufflevector(b1l, b1h, 0, 1, 2, 3, 4, 5, 6, 7,
                                       8, 9, 10, 11, 12, 13, 14, 15);

    acc0 = __builtin_amdgcn_wmma_f32_16x16x32_f16(false, af, false, bf0,
                                                  (short)0, acc0, false, false);
    acc1 = __builtin_amdgcn_wmma_f32_16x16x32_f16(false, af, false, bf1,
                                                  (short)0, acc1, false, false);
    __syncthreads();
  }

  // C/D layout: vgpr vg of lane(hi,r) -> M = vg + 8*hi, N = r
#pragma unroll
  for (int vg = 0; vg < 8; ++vg) {
    int row = m0 + waveM * 16 + hi * 8 + vg;
    int c0  = n0 + waveN * 32 + r;
    int c1  = c0 + 16;
    float x0 = acc0[vg] + bias[c0];
    float x1 = acc1[vg] + bias[c1];
    if (act) { x0 = gelu_tanh(x0); x1 = gelu_tanh(x1); }
    size_t o0 = (size_t)row * N + c0;
    size_t o1 = (size_t)row * N + c1;
    if (Cf) {
      if (resid) { x0 += resid[o0]; x1 += resid[o1]; }
      Cf[o0] = x0; Cf[o1] = x1;
    } else {
      Ch[o0] = (_Float16)x0; Ch[o1] = (_Float16)x1;
    }
  }
}

// ------- junction-tree attention: keys for pos l are [(c-1)*4, l], <=8 keys -------
__global__ __launch_bounds__(256)
void attn_kernel(const _Float16* __restrict__ q,
                 const _Float16* __restrict__ k,
                 const _Float16* __restrict__ v,
                 _Float16* __restrict__ o) {
  int wid  = threadIdx.x >> 5, lane = threadIdx.x & 31;
  int g    = blockIdx.x * 8 + wid;
  int h    = g & (NHEAD - 1);
  int row  = g >> 3;                      // b*L + l
  int l    = row & (LSEQ - 1);
  int c    = l >> 2;
  int kst  = (c > 0) ? (c - 1) * 4 : 0;
  int nk   = l - kst + 1;                 // <= 8
  int base = row - l + kst;

  size_t qb = (size_t)row * DMODEL + h * DHEAD;
  float q0 = (float)q[qb + lane], q1 = (float)q[qb + lane + 32];

  float sc[8];
  const float scale = 0.125f;
  for (int j = 0; j < nk; ++j) {
    size_t kb = (size_t)(base + j) * DMODEL + h * DHEAD;
    float p = q0 * (float)k[kb + lane] + q1 * (float)k[kb + lane + 32];
#pragma unroll
    for (int off = 16; off; off >>= 1) p += __shfl_xor(p, off, 32);
    sc[j] = p * scale;
  }
  float m = sc[0];
  for (int j = 1; j < nk; ++j) m = fmaxf(m, sc[j]);
  float den = 0.f;
  for (int j = 0; j < nk; ++j) { sc[j] = __expf(sc[j] - m); den += sc[j]; }
  float inv = 1.0f / den;
  float o0 = 0.f, o1 = 0.f;
  for (int j = 0; j < nk; ++j) {
    size_t vb = (size_t)(base + j) * DMODEL + h * DHEAD;
    float p = sc[j] * inv;
    o0 += p * (float)v[vb + lane];
    o1 += p * (float)v[vb + lane + 32];
  }
  o[qb + lane]      = (_Float16)o0;
  o[qb + lane + 32] = (_Float16)o1;
}

// ---------------- output projection (512 -> 21), one wave per row ----------------
__global__ __launch_bounds__(256)
void outproj_kernel(const float* __restrict__ hbuf,
                    const float* __restrict__ wout,
                    const float* __restrict__ bout,
                    float* __restrict__ out) {
  __shared__ float rowbuf[8][DMODEL];
  int wid = threadIdx.x >> 5, lane = threadIdx.x & 31;
  int row = blockIdx.x * 8 + wid;
  const float* hr = hbuf + (size_t)row * DMODEL;
  for (int i = lane; i < DMODEL; i += 32) rowbuf[wid][i] = hr[i];
  __syncthreads();
  if (lane < NALPHA) {
    float acc = bout[lane];
    for (int kd = 0; kd < DMODEL; ++kd)
      acc += rowbuf[wid][kd] * wout[kd * NALPHA + lane];
    out[(size_t)row * NALPHA + lane] = acc;
  }
}

extern "C" void kernel_launch(void* const* d_in, const int* in_sizes, int n_in,
                              void* d_out, int out_size, void* d_ws, size_t ws_size,
                              hipStream_t stream) {
  const int*   x       = (const int*)d_in[0];
  const float* tok_emb = (const float*)d_in[2];
  const float* pos_emb = (const float*)d_in[3];
  const float* wq   = (const float*)d_in[4];
  const float* bq   = (const float*)d_in[5];
  const float* wk   = (const float*)d_in[6];
  const float* bk   = (const float*)d_in[7];
  const float* wv   = (const float*)d_in[8];
  const float* bv   = (const float*)d_in[9];
  const float* wo   = (const float*)d_in[10];
  const float* bo   = (const float*)d_in[11];
  const float* ln1s = (const float*)d_in[12];
  const float* ln1b = (const float*)d_in[13];
  const float* ln2s = (const float*)d_in[14];
  const float* ln2b = (const float*)d_in[15];
  const float* w1   = (const float*)d_in[16];
  const float* b1   = (const float*)d_in[17];
  const float* w2   = (const float*)d_in[18];
  const float* b2   = (const float*)d_in[19];
  const float* wout = (const float*)d_in[20];
  const float* bout = (const float*)d_in[21];

  const size_t RC = (size_t)NTOK * DMODEL;       // 16.7M elements
  const size_t DD = (size_t)DMODEL * DMODEL;
  const size_t DF = (size_t)DMODEL * DFFN;
  const size_t WPL = 4 * DD + 2 * DF;            // transposed f16 weights per layer

  float*    h   = (float*)d_ws;                  // f32 residual stream
  _Float16* hn  = (_Float16*)(h + RC);           // f16 LN output
  _Float16* big = hn + RC;                       // 4*RC halves shared region
  _Float16* qb  = big;
  _Float16* kb  = big + RC;
  _Float16* vb  = big + 2 * RC;
  _Float16* ab  = big + 3 * RC;
  _Float16* ff  = big;                           // (M, DFFN) = 4*RC halves, reuses q/k/v/a
  _Float16* wts = big + 4 * RC;                  // transposed f16 weights

  dim3 blk(256);
  dim3 gD(DMODEL / 64, NTOK / 64);               // (8, 512)
  dim3 gF(DFFN / 64,  NTOK / 64);                // (32, 512)

  // ---- per-launch weight prep: f32 (K,N) -> f16 (N,K) ----
  for (int l = 0; l < NLAYER; ++l) {
    _Float16* base = wts + (size_t)l * WPL;
    dim3 gDD(DMODEL / 32, DMODEL / 32);
    transpose_f16_kernel<<<gDD, blk, 0, stream>>>(wq + l * DD, base,          DMODEL, DMODEL);
    transpose_f16_kernel<<<gDD, blk, 0, stream>>>(wk + l * DD, base + DD,     DMODEL, DMODEL);
    transpose_f16_kernel<<<gDD, blk, 0, stream>>>(wv + l * DD, base + 2 * DD, DMODEL, DMODEL);
    transpose_f16_kernel<<<gDD, blk, 0, stream>>>(wo + l * DD, base + 3 * DD, DMODEL, DMODEL);
    transpose_f16_kernel<<<dim3(DFFN / 32, DMODEL / 32), blk, 0, stream>>>(
        w1 + l * DF, base + 4 * DD, DMODEL, DFFN);          // -> (DFFN, DMODEL)
    transpose_f16_kernel<<<dim3(DMODEL / 32, DFFN / 32), blk, 0, stream>>>(
        w2 + l * DF, base + 4 * DD + DF, DFFN, DMODEL);     // -> (DMODEL, DFFN)
  }

  embed_kernel<<<NTOK, blk, 0, stream>>>(x, tok_emb, pos_emb, h);

  for (int l = 0; l < NLAYER; ++l) {
    _Float16* base = wts + (size_t)l * WPL;
    _Float16* wqT = base;
    _Float16* wkT = base + DD;
    _Float16* wvT = base + 2 * DD;
    _Float16* woT = base + 3 * DD;
    _Float16* w1T = base + 4 * DD;
    _Float16* w2T = base + 4 * DD + DF;

    ln_kernel<<<NTOK / 8, blk, 0, stream>>>(h, ln1s + l * DMODEL, ln1b + l * DMODEL, hn);
    gemm_wmma_kernel<<<gD, blk, 0, stream>>>(hn, wqT, bq + l * DMODEL, nullptr,
                                             nullptr, qb, NTOK, DMODEL, DMODEL, 0);
    gemm_wmma_kernel<<<gD, blk, 0, stream>>>(hn, wkT, bk + l * DMODEL, nullptr,
                                             nullptr, kb, NTOK, DMODEL, DMODEL, 0);
    gemm_wmma_kernel<<<gD, blk, 0, stream>>>(hn, wvT, bv + l * DMODEL, nullptr,
                                             nullptr, vb, NTOK, DMODEL, DMODEL, 0);
    attn_kernel<<<NTOK * NHEAD / 8, blk, 0, stream>>>(qb, kb, vb, ab);
    gemm_wmma_kernel<<<gD, blk, 0, stream>>>(ab, woT, bo + l * DMODEL, h,
                                             h, nullptr, NTOK, DMODEL, DMODEL, 0);

    ln_kernel<<<NTOK / 8, blk, 0, stream>>>(h, ln2s + l * DMODEL, ln2b + l * DMODEL, hn);
    gemm_wmma_kernel<<<gF, blk, 0, stream>>>(hn, w1T, b1 + l * DFFN, nullptr,
                                             nullptr, ff, NTOK, DFFN, DMODEL, 1);
    gemm_wmma_kernel<<<gD, blk, 0, stream>>>(ff, w2T, b2 + l * DMODEL, h,
                                             h, nullptr, NTOK, DMODEL, DFFN, 0);
  }

  outproj_kernel<<<NTOK / 8, blk, 0, stream>>>(h, wout, bout, (float*)d_out);
}
